// HierarchicalLocalAttentionND_39754217292521
// MI455X (gfx1250) — matrix operand order repaired and checked by reading, not compiled, not run on hardware
//
#include <hip/hip_runtime.h>
#include <hip/hip_bf16.h>
#include <math.h>

typedef __attribute__((ext_vector_type(16))) _Float16 v16h;
typedef __attribute__((ext_vector_type(8)))  _Float16 v8h;
typedef __attribute__((ext_vector_type(8)))  float    v8f;

#define BB    4
#define LL    4096
#define ED    512
#define NH    8
#define HD    64
#define MTOT  (BB * LL)        // 16384
#define NPAD  1600             // 1536 qkv + 16 wp + 48 zero pad
#define SCALE 0.125f           // 64^-0.5
#define EPS   1e-6f

// ---------------- fragment helpers ----------------
__device__ inline v16h zero16() {
  v16h z;
#pragma unroll
  for (int i = 0; i < 16; ++i) z[i] = (_Float16)0.0f;
  return z;
}
__device__ inline v8f zero8f() {
  v8f z;
#pragma unroll
  for (int i = 0; i < 8; ++i) z[i] = 0.0f;
  return z;
}
// A 16x32 f16 fragment: lane half h holds 16 contiguous K = h*16..h*16+15.
// p = row_ptr + k0 + h*16.
__device__ inline v16h load_a_frag(const _Float16* p) {
  v8h lo = *(const v8h*)p;
  v8h hi = *(const v8h*)(p + 8);
  v16h f;
#pragma unroll
  for (int i = 0; i < 8; ++i) { f[i] = lo[i]; f[i + 8] = hi[i]; }
  return f;
}
// B 32x16 f16 fragment: lane half h holds 16 contiguous K = h*16..h*16+15.
// p = col_ptr + k0 + h*16.
__device__ inline v16h load_b_frag(const _Float16* p) {
  v8h lo = *(const v8h*)p;
  v8h hi = *(const v8h*)(p + 8);
  v16h f;
#pragma unroll
  for (int i = 0; i < 8; ++i) { f[i] = lo[i]; f[i + 8] = hi[i]; }
  return f;
}

// ---------------- conversion kernels ----------------
__global__ void cvt_f32_f16(const float* __restrict__ src,
                            _Float16* __restrict__ dst, size_t n) {
  size_t i = (size_t)blockIdx.x * 256 + threadIdx.x;
  if (i < n) dst[i] = (_Float16)src[i];
}

__global__ void build_wA(const float* __restrict__ w_qkv,
                         const float* __restrict__ w_wp,
                         _Float16* __restrict__ dst) {
  size_t i = (size_t)blockIdx.x * 256 + threadIdx.x;
  if (i >= (size_t)NPAD * ED) return;
  int row = (int)(i / ED), col = (int)(i % ED);
  float v = 0.0f;
  if (row < 1536)      v = w_qkv[i];
  else if (row < 1552) v = w_wp[(size_t)(row - 1536) * ED + col];
  dst[i] = (_Float16)v;
}

// ---------------- f16 WMMA GEMM: D[M,N] = A[M,K] * Bw[N,K]^T ----------------
// one wave computes a 32(M) x 64(N) tile; K%32==0, M%32==0, N%64==0
__global__ __launch_bounds__(32) void gemm_f16_wmma(
    const _Float16* __restrict__ A, const _Float16* __restrict__ Bw,
    float* __restrict__ D, int M, int N, int K) {
  const int lane = threadIdx.x;
  const int idx = lane & 15, half = lane >> 4;
  const int n0 = blockIdx.x * 64;
  const int m0 = blockIdx.y * 32;

  v8f acc[2][4];
#pragma unroll
  for (int mt = 0; mt < 2; ++mt)
#pragma unroll
    for (int nt = 0; nt < 4; ++nt) acc[mt][nt] = zero8f();

  for (int k0 = 0; k0 < K; k0 += 32) {
    v16h a[2], bf[4];
#pragma unroll
    for (int mt = 0; mt < 2; ++mt)
      a[mt] = load_a_frag(A + (size_t)(m0 + mt * 16 + idx) * K + k0 + half * 16);
#pragma unroll
    for (int nt = 0; nt < 4; ++nt)
      bf[nt] = load_b_frag(Bw + (size_t)(n0 + nt * 16 + idx) * K + k0 + half * 16);
    if (k0 + 32 < K) {  // L2 prefetch of next K slab
      __builtin_prefetch(A + (size_t)(m0 + idx) * K + k0 + 32, 0, 1);
      __builtin_prefetch(Bw + (size_t)(n0 + idx) * K + k0 + 32, 0, 1);
    }
#pragma unroll
    for (int mt = 0; mt < 2; ++mt)
#pragma unroll
      for (int nt = 0; nt < 4; ++nt)
        acc[mt][nt] = __builtin_amdgcn_wmma_f32_16x16x32_f16(
            false, a[mt], false, bf[nt], (short)0, acc[mt][nt], false, false);
  }
#pragma unroll
  for (int mt = 0; mt < 2; ++mt)
#pragma unroll
    for (int nt = 0; nt < 4; ++nt)
#pragma unroll
      for (int v = 0; v < 8; ++v)
        D[(size_t)(m0 + mt * 16 + v + 8 * half) * N + n0 + nt * 16 + idx] =
            acc[mt][nt][v];
}

// ---------------- per-token prep: rmsnorm + rope + softplus/sigmoid --------
__global__ __launch_bounds__(64) void prep_qkv(
    const float* __restrict__ qkv, const float* __restrict__ qn_w,
    const float* __restrict__ kn_w, const float* __restrict__ b_wp,
    _Float16* __restrict__ q_r, _Float16* __restrict__ k_r,
    _Float16* __restrict__ vT, float* __restrict__ decay,
    float* __restrict__ gate) {
  __shared__ float red[64];
  __shared__ float qb[64], kb[64];
  const int t = blockIdx.x;
  const int b = t / LL, l = t % LL;
  const int d = threadIdx.x;
  const float* row = qkv + (size_t)t * NPAD;

  const int j = d & 31;  // rope frequency index
  const float freq = __expf(-(float)j * (9.210340371976184f / 32.0f));
  const float ang = (float)l * freq;
  const float cs = __cosf(ang), sn = __sinf(ang);

  for (int h = 0; h < NH; ++h) {
    float qv = row[h * HD + d];
    float kv = row[ED + h * HD + d];
    float vv = row[2 * ED + h * HD + d];

    red[d] = qv * qv; __syncthreads();
    for (int off = 32; off > 0; off >>= 1) {
      if (d < off) red[d] += red[d + off];
      __syncthreads();
    }
    float qss = red[0]; __syncthreads();
    float qn = qv * __frsqrt_rn(qss * (1.0f / 64.0f) + EPS) * qn_w[d];

    red[d] = kv * kv; __syncthreads();
    for (int off = 32; off > 0; off >>= 1) {
      if (d < off) red[d] += red[d + off];
      __syncthreads();
    }
    float kss = red[0]; __syncthreads();
    float kn = kv * __frsqrt_rn(kss * (1.0f / 64.0f) + EPS) * kn_w[d];

    qb[d] = qn; kb[d] = kn; __syncthreads();
    float qo, ko;
    if (d < 32) {
      qo = qb[d] * cs - qb[d + 32] * sn;
      ko = kb[d] * cs - kb[d + 32] * sn;
    } else {
      qo = qb[d - 32] * sn + qb[d] * cs;
      ko = kb[d - 32] * sn + kb[d] * cs;
    }
    __syncthreads();

    size_t base = ((size_t)(b * NH + h) * LL + l) * HD + d;
    q_r[base] = (_Float16)qo;
    k_r[base] = (_Float16)ko;
    vT[((size_t)(b * NH + h) * HD + d) * LL + l] = (_Float16)vv;
  }

  if (d < NH) {
    float w0 = row[1536 + 2 * d] + b_wp[2 * d];
    float w1 = row[1536 + 2 * d + 1] + b_wp[2 * d + 1];
    float dc = (w0 > 20.0f) ? w0 : log1pf(__expf(w0));
    float gt = 1.0f / (1.0f + __expf(-w1));
    decay[(size_t)(b * NH + d) * LL + l] = dc;
    gate[(size_t)(b * NH + d) * LL + l] = gt;
  }
}

// ---------------- windowed attention: one wave per (b,h,16-query tile) -----
__global__ __launch_bounds__(32) void attn_wmma(
    const _Float16* __restrict__ q_r, const _Float16* __restrict__ k_r,
    const _Float16* __restrict__ vT, const float* __restrict__ decay,
    const float* __restrict__ gate, float* __restrict__ out) {
  __shared__ __align__(16) _Float16 plds[16 * 64];
  const int lane = threadIdx.x;
  const int idx = lane & 15, half = lane >> 4;
  const int qt = blockIdx.x % (LL / 16);
  const int h = (blockIdx.x / (LL / 16)) % NH;
  const int b = blockIdx.x / ((LL / 16) * NH);
  const int qbase = qt * 16;
  const size_t headoff = (size_t)(b * NH + h) * LL;

  // q A-fragments, K(dim)=64 -> 2 chunks of 32
  v16h aq[2];
#pragma unroll
  for (int ks = 0; ks < 2; ++ks)
    aq[ks] = load_a_frag(q_r + (headoff + qbase + idx) * HD + ks * 32 + half * 16);

  // QK^T over 48-key union window (3 tiles of 16)
  v8f lacc[3];
#pragma unroll
  for (int t = 0; t < 3; ++t) lacc[t] = zero8f();
#pragma unroll
  for (int t = 0; t < 3; ++t) {
    const int tok = qbase - 16 + t * 16 + idx;
    v16h bk[2];
    if (tok >= 0 && tok < LL) {
#pragma unroll
      for (int ks = 0; ks < 2; ++ks)
        bk[ks] = load_b_frag(k_r + (headoff + tok) * HD + ks * 32 + half * 16);
    } else {
      bk[0] = zero16(); bk[1] = zero16();  // reference zero-pads k
    }
#pragma unroll
    for (int ks = 0; ks < 2; ++ks)
      lacc[t] = __builtin_amdgcn_wmma_f32_16x16x32_f16(
          false, aq[ks], false, bk[ks], (short)0, lacc[t], false, false);
  }

  float dec[8], g[8];
#pragma unroll
  for (int v = 0; v < 8; ++v) {
    const int m = v + 8 * half;
    dec[v] = decay[headoff + qbase + m];
    g[v] = gate[headoff + qbase + m];
  }

  // masked logits + 16-lane butterfly softmax (row m lives on 16 lanes x 3 tiles)
  float lg[3][8];
#pragma unroll
  for (int t = 0; t < 3; ++t)
#pragma unroll
    for (int v = 0; v < 8; ++v) {
      const int m = v + 8 * half;
      const int win = t * 16 + idx - m;  // window position 0..32 valid
      const float rel = fabsf((float)(win - 16));
      lg[t][v] = (win >= 0 && win <= 32)
                     ? (lacc[t][v] * SCALE - dec[v] * rel)
                     : -1e30f;
    }
#pragma unroll
  for (int v = 0; v < 8; ++v) {
    float mx = fmaxf(fmaxf(lg[0][v], lg[1][v]), lg[2][v]);
#pragma unroll
    for (int o = 1; o < 16; o <<= 1) mx = fmaxf(mx, __shfl_xor(mx, o, 32));
    float e = __expf(lg[0][v] - mx) + __expf(lg[1][v] - mx) + __expf(lg[2][v] - mx);
#pragma unroll
    for (int o = 1; o < 16; o <<= 1) e += __shfl_xor(e, o, 32);
    const float inv = 1.0f / e;
    const int m = v + 8 * half;
#pragma unroll
    for (int t = 0; t < 3; ++t)
      plds[m * 64 + t * 16 + idx] = (_Float16)(__expf(lg[t][v] - mx) * inv);
    plds[m * 64 + 48 + idx] = (_Float16)0.0f;  // pad keys 48..63
  }
  __syncthreads();

  // probs as A-fragments (K=keys, padded to 64)
  v16h ap[2];
#pragma unroll
  for (int ks = 0; ks < 2; ++ks)
    ap[ks] = load_a_frag(plds + idx * 64 + ks * 32 + half * 16);

  // A*V with pre-transposed V: vT[b,h][dim][token]
  v8f oacc[4];
#pragma unroll
  for (int nt = 0; nt < 4; ++nt) oacc[nt] = zero8f();
  const _Float16* vbase = vT + (size_t)(b * NH + h) * HD * LL;
#pragma unroll
  for (int ks = 0; ks < 2; ++ks) {
#pragma unroll
    for (int nt = 0; nt < 4; ++nt) {
      const int dim = nt * 16 + idx;
      const int key0 = qbase - 16 + ks * 32 + half * 16;
      const _Float16* p = vbase + (size_t)dim * LL + key0;
      v16h bv;
      if (key0 >= 0 && key0 + 15 < LL) {
        bv = load_b_frag(p);
      } else {
#pragma unroll
        for (int i = 0; i < 16; ++i) {
          const int kk = key0 + i;
          bv[i] = (kk >= 0 && kk < LL) ? p[i] : (_Float16)0.0f;
        }
      }
      oacc[nt] = __builtin_amdgcn_wmma_f32_16x16x32_f16(
          false, ap[ks], false, bv, (short)0, oacc[nt], false, false);
    }
  }

#pragma unroll
  for (int nt = 0; nt < 4; ++nt)
#pragma unroll
    for (int v = 0; v < 8; ++v) {
      const int m = v + 8 * half;
      out[((size_t)b * LL + qbase + m) * ED + h * HD + nt * 16 + idx] =
          oacc[nt][v] * g[v];
    }
}

// ---------------- final rmsnorm (over 512) + f16 convert -------------------
__global__ __launch_bounds__(256) void rmsnorm_out(const float* __restrict__ x,
                                                   const float* __restrict__ w,
                                                   _Float16* __restrict__ y) {
  __shared__ float red[256];
  const int t = blockIdx.x, tid = threadIdx.x;
  const float* row = x + (size_t)t * ED;
  const float a = row[tid], b2 = row[tid + 256];
  red[tid] = a * a + b2 * b2;
  __syncthreads();
  for (int off = 128; off > 0; off >>= 1) {
    if (tid < off) red[tid] += red[tid + off];
    __syncthreads();
  }
  const float inv = __frsqrt_rn(red[0] * (1.0f / 512.0f) + EPS);
  y[(size_t)t * ED + tid] = (_Float16)(a * inv * w[tid]);
  y[(size_t)t * ED + tid + 256] = (_Float16)(b2 * inv * w[tid + 256]);
}

// ---------------- host orchestration ----------------
extern "C" void kernel_launch(void* const* d_in, const int* in_sizes, int n_in,
                              void* d_out, int out_size, void* d_ws,
                              size_t ws_size, hipStream_t stream) {
  (void)in_sizes; (void)n_in; (void)out_size; (void)ws_size;
  const float* x     = (const float*)d_in[0];
  const float* w_qkv = (const float*)d_in[1];
  const float* qn_w  = (const float*)d_in[2];
  const float* kn_w  = (const float*)d_in[3];
  const float* w_wp  = (const float*)d_in[4];
  const float* b_wp  = (const float*)d_in[5];
  const float* on_w  = (const float*)d_in[6];
  const float* w_out = (const float*)d_in[7];
  float* out = (float*)d_out;

  char* ws = (char*)d_ws;
  size_t off = 0;
  auto alloc = [&](size_t bytes) -> void* {
    void* p = ws + off;
    off += (bytes + 255) & ~(size_t)255;
    return p;
  };
  _Float16* xh     = (_Float16*)alloc((size_t)MTOT * ED * 2);
  _Float16* wA     = (_Float16*)alloc((size_t)NPAD * ED * 2);
  _Float16* wO     = (_Float16*)alloc((size_t)ED * ED * 2);
  float*    qkv    = (float*)   alloc((size_t)MTOT * NPAD * 4);
  _Float16* q_r    = (_Float16*)alloc((size_t)MTOT * ED * 2);
  _Float16* k_r    = (_Float16*)alloc((size_t)MTOT * ED * 2);
  _Float16* vT     = (_Float16*)alloc((size_t)MTOT * ED * 2);
  float*    decay  = (float*)   alloc((size_t)BB * NH * LL * 4);
  float*    gate   = (float*)   alloc((size_t)BB * NH * LL * 4);
  float*    attn_o = (float*)   alloc((size_t)MTOT * ED * 4);
  _Float16* normed = (_Float16*)alloc((size_t)MTOT * ED * 2);

  const size_t nx = (size_t)MTOT * ED;
  cvt_f32_f16<<<dim3((unsigned)((nx + 255) / 256)), dim3(256), 0, stream>>>(x, xh, nx);
  build_wA<<<dim3((unsigned)(((size_t)NPAD * ED + 255) / 256)), dim3(256), 0, stream>>>(
      w_qkv, w_wp, wA);
  const size_t nwo = (size_t)ED * ED;
  cvt_f32_f16<<<dim3((unsigned)((nwo + 255) / 256)), dim3(256), 0, stream>>>(w_out, wO, nwo);

  gemm_f16_wmma<<<dim3(NPAD / 64, MTOT / 32), dim3(32), 0, stream>>>(
      xh, wA, qkv, MTOT, NPAD, ED);

  prep_qkv<<<dim3(MTOT), dim3(64), 0, stream>>>(qkv, qn_w, kn_w, b_wp, q_r, k_r,
                                                vT, decay, gate);

  attn_wmma<<<dim3(BB * NH * (LL / 16)), dim3(32), 0, stream>>>(
      q_r, k_r, vT, decay, gate, attn_o);

  rmsnorm_out<<<dim3(MTOT), dim3(256), 0, stream>>>(attn_o, on_w, normed);

  gemm_f16_wmma<<<dim3(ED / 64, MTOT / 32), dim3(32), 0, stream>>>(
      normed, wO, out, MTOT, ED, ED);
}
